// RCCAM_CAM_Layer_64115271794850
// MI455X (gfx1250) — matrix operand order repaired and checked by reading, not compiled, not run on hardware
//
#include <hip/hip_runtime.h>

typedef __attribute__((ext_vector_type(16))) __bf16 v16bf;
typedef __attribute__((ext_vector_type(8)))  __bf16 v8bf;
typedef __attribute__((ext_vector_type(8)))  float  v8f;

#define HH 96
#define WW 96
#define NB 4
#define LSTRIDE 56   // LDS row stride (elements): 112B -> 16B aligned, bank-conflict-free

// ---------- bf16 helpers ----------
__device__ inline float bf2f(__bf16 x){
  unsigned short u = __builtin_bit_cast(unsigned short, x);
  unsigned int v = ((unsigned int)u) << 16;
  return __builtin_bit_cast(float, v);
}
__device__ inline __bf16 f2bf(float f){
  unsigned int v = __builtin_bit_cast(unsigned int, f);
  unsigned int r = v + 0x7FFFu + ((v >> 16) & 1u);   // RNE
  return __builtin_bit_cast(__bf16, (unsigned short)(r >> 16));
}

// ---------- CDNA5 async global->LDS DMA (bypasses VGPRs, tracked by ASYNCcnt) ----------
__device__ inline void async_b128(__bf16* lds, const __bf16* g){
  unsigned loff = (unsigned)(unsigned long long)(void*)lds;      // LDS aperture: addr[31:0] = offset
  asm volatile("global_load_async_to_lds_b128 %0, %1, off"
               :: "v"(loff), "v"((unsigned long long)(const void*)g) : "memory");
}
__device__ inline void wait_async0(){
  asm volatile("s_wait_asynccnt 0x0" ::: "memory");
}

// Build a 16-element fragment from two aligned 16B LDS runs.
__device__ inline v16bf frag16(const __bf16* p0, const __bf16* p1){
  v8bf a = *(const v8bf*)p0;
  v8bf b = *(const v8bf*)p1;
  v16bf r;
#pragma unroll
  for (int i = 0; i < 8; ++i){ r[i] = a[i]; r[i+8] = b[i]; }
  return r;
}

// ---------- elementwise fp32 -> bf16 ----------
__global__ void k_cvt_bf16(const float* __restrict__ in, unsigned short* __restrict__ out, int n){
  int i = blockIdx.x*256 + threadIdx.x;
  if (i < n) out[i] = __builtin_bit_cast(unsigned short, f2bf(in[i]));
}

// ---------- fold BN into per-channel scale/bias ----------
__global__ void k_bn_fuse(const float* s, const float* b, const float* m, const float* v,
                          float* scale, float* bias, int c){
  int i = blockIdx.x*256 + threadIdx.x;
  if (i < c){
    float inv = s[i] * rsqrtf(v[i] + 1e-5f);
    scale[i] = inv;
    bias[i]  = b[i] - m[i]*inv;
  }
}

// ---------- implicit-GEMM conv (KS=1 or 3), bf16 WMMA, fp32 accumulate ----------
// Block tile: 128(M) x 128(N); N tile = 4 image rows x 32 x. 8 waves (4Mx2N), each 32x64:
// 8 WMMA per k-step from 2 A-frags + 4 B-frags (12 ds_load_b128).
// A [M][K] row-major bf16 staged by async DMA (double buffered); B staged via registers (transpose+pad).
template<int KS>
__global__ __launch_bounds__(256) void k_conv_wmma(
    const unsigned short* __restrict__ in1r, int C1,
    const unsigned short* __restrict__ in2r, int C2,
    const unsigned short* __restrict__ Ar, long a_bstride,
    int M, int K,
    const float* __restrict__ scale, const float* __restrict__ bias, int relu,
    unsigned short* __restrict__ out_bf, float* __restrict__ out_f)
{
  constexpr int KK  = KS*KS;
  constexpr int PAD = KS/2;
  __shared__ __bf16 Asm[2*128*LSTRIDE];
  __shared__ __bf16 Bsm[2*128*LSTRIDE];

  const __bf16* in1 = (const __bf16*)in1r;
  const __bf16* in2 = (const __bf16*)in2r;

  const int tid = threadIdx.x;
  const int nt  = blockIdx.x;
  const int xb_ = nt % 3;
  const int yq  = (nt/3) % (HH/4);
  const int b   = nt / (3*(HH/4));
  const int x0  = xb_*32;
  const int y0  = yq*4;
  const int m0  = blockIdx.y*128;

  const __bf16* A = (const __bf16*)Ar + (long)b*a_bstride;

  v8f acc[2][4] = {};

  const int lane = tid & 31;
  const int wv   = tid >> 5;
  const int mg   = wv & 3;          // 4 M-groups of 32 rows
  const int ng   = wv >> 2;         // 2 N-groups of 64 cols (= image rows y0+2ng .. y0+2ng+1)
  const int lh   = (lane < 16) ? 0 : 1;
  const int lm   = lane & 15;

  // B staging: thread covers 16 consecutive n of one image row for one k
  const int kl    = tid >> 3;          // k lane 0..31
  const int g8    = tid & 7;           // n-group
  const int ryS   = g8 >> 1;           // image row within tile (0..3)
  const int nbase = g8*16;             // n_local base (0..112)
  const int xbase = x0 + (g8 & 1)*16;  // x base

  const int ksteps = K >> 5;
  unsigned short breg[16];

  auto stageA = [&](int kt, int buf){
#pragma unroll
    for (int i = 0; i < 2; ++i){
      int chunk = tid*2 + i;             // 0..511
      int row = chunk >> 2, cb = (chunk & 3)*8;
      int gm = m0 + row; if (gm >= M) gm = M - 1;   // clamp: rows >= M never stored
      async_b128(&Asm[buf*128*LSTRIDE + row*LSTRIDE + cb],
                 A + (long)gm*K + kt*32 + cb);
    }
  };
  auto loadB = [&](int kt){
    int k  = kt*32 + kl;
    int ic = k / KK;
    int r  = k - ic*KK;
    int ky = (KS == 3) ? (r/3)      : 0;
    int kx = (KS == 3) ? (r - ky*3) : 0;
    int ys = y0 + ryS + ky - PAD;
    const __bf16* src = (ic < C1)
        ? in1 + ((long)(b*C1 + ic)*HH + ys)*WW
        : in2 + ((long)(b*C2 + (ic - C1))*HH + ys)*WW;
    bool yok = (ys >= 0) && (ys < HH);
#pragma unroll
    for (int j = 0; j < 16; ++j){
      int xs = xbase + j + kx - PAD;
      unsigned short v = 0;
      if (yok && xs >= 0 && xs < WW) v = __builtin_bit_cast(unsigned short, src[xs]);
      breg[j] = v;
    }
  };
  auto storeB = [&](int buf){
#pragma unroll
    for (int j = 0; j < 16; ++j)
      Bsm[buf*128*LSTRIDE + (nbase + j)*LSTRIDE + kl] = __builtin_bit_cast(__bf16, breg[j]);
  };
  auto compute = [&](int buf){
    const __bf16* Ab = &Asm[buf*128*LSTRIDE];
    const __bf16* Bb = &Bsm[buf*128*LSTRIDE];
    v16bf bfr[4];
#pragma unroll
    for (int sn = 0; sn < 4; ++sn){
      const __bf16* pb = &Bb[(ng*64 + sn*16 + lm)*LSTRIDE + lh*8];
      bfr[sn] = frag16(pb, pb + 16);
    }
#pragma unroll
    for (int sm = 0; sm < 2; ++sm){
      const __bf16* pa = &Ab[(mg*32 + sm*16 + lm)*LSTRIDE + lh*8];
      v16bf af = frag16(pa, pa + 16);
#pragma unroll
      for (int sn = 0; sn < 4; ++sn)
        acc[sm][sn] = __builtin_amdgcn_wmma_f32_16x16x32_bf16(false, af, false, bfr[sn],
                                                              (short)0, acc[sm][sn], false, false);
    }
  };

  // prologue
  stageA(0, 0); loadB(0); storeB(0);
  wait_async0();
  __syncthreads();

  int cur = 0;
  for (int kt = 0; kt < ksteps; ++kt){
    int nxt = cur ^ 1;
    bool more = (kt + 1) < ksteps;
    if (more){ stageA(kt + 1, nxt); loadB(kt + 1); }   // DMA + global loads in flight...
    compute(cur);                                      // ...while WMMAs run
    if (more){ storeB(nxt); }
    wait_async0();
    __syncthreads();
    cur = nxt;
  }

  // epilogue
#pragma unroll
  for (int sm = 0; sm < 2; ++sm){
#pragma unroll
    for (int sn = 0; sn < 4; ++sn){
      int yO = y0 + 2*ng + (sn >> 1);
      int xO = x0 + (sn & 1)*16 + lm;
#pragma unroll
      for (int j = 0; j < 8; ++j){
        int m = m0 + mg*32 + sm*16 + lh*8 + j;
        if (m < M){
          float v = acc[sm][sn][j];
          if (scale) v *= scale[m];
          if (bias)  v += bias[m];
          if (relu)  v = v > 0.f ? v : 0.f;
          long oi = ((long)(b*M + m)*HH + yO)*WW + xO;
          if (out_f) out_f[oi] = v;
          else       out_bf[oi] = __builtin_bit_cast(unsigned short, f2bf(v));
        }
      }
    }
  }
}

// ---------- CAM energy: E[b] = X[b] * X[b]^T (C x C, K = H*W); fully async-staged ----------
__global__ __launch_bounds__(256) void k_gemm_abt(
    const unsigned short* __restrict__ Xr, float* __restrict__ E, int C, int Kd)
{
  __shared__ __bf16 Asm[2*128*LSTRIDE];
  __shared__ __bf16 Bsm[2*64*LSTRIDE];
  const __bf16* X = (const __bf16*)Xr;
  const int tid = threadIdx.x;
  const int b  = blockIdx.z;
  const int n0 = blockIdx.x*64;
  const int m0 = blockIdx.y*128;
  const __bf16* Xb = X + (long)b*C*Kd;
  v8f acc[2][2] = {};
  const int lane = tid & 31, wv = tid >> 5, mg = wv & 3, ng = wv >> 2;
  const int lh = (lane < 16) ? 0 : 1, lm = lane & 15;
  const int bn = tid >> 2;            // 0..63
  const int bk = (tid & 3)*8;         // 8 consecutive k

  auto stage = [&](int kt, int buf){
#pragma unroll
    for (int i = 0; i < 2; ++i){
      int chunk = tid*2 + i; int row = chunk >> 2; int cb = (chunk & 3)*8;
      async_b128(&Asm[buf*128*LSTRIDE + row*LSTRIDE + cb],
                 Xb + (long)(m0 + row)*Kd + kt*32 + cb);
    }
    async_b128(&Bsm[buf*64*LSTRIDE + bn*LSTRIDE + bk],
               Xb + (long)(n0 + bn)*Kd + kt*32 + bk);
  };
  auto compute = [&](int buf){
    const __bf16* Ab = &Asm[buf*128*LSTRIDE];
    const __bf16* Bb = &Bsm[buf*64*LSTRIDE];
    const __bf16* p0 = &Bb[(ng*32 + lm)*LSTRIDE + lh*8];
    v16bf bf0 = frag16(p0, p0 + 16);
    const __bf16* p1 = &Bb[(ng*32 + 16 + lm)*LSTRIDE + lh*8];
    v16bf bf1 = frag16(p1, p1 + 16);
#pragma unroll
    for (int sm = 0; sm < 2; ++sm){
      const __bf16* pa = &Ab[(mg*32 + sm*16 + lm)*LSTRIDE + lh*8];
      v16bf af = frag16(pa, pa + 16);
      acc[sm][0] = __builtin_amdgcn_wmma_f32_16x16x32_bf16(false, af, false, bf0,
                                                           (short)0, acc[sm][0], false, false);
      acc[sm][1] = __builtin_amdgcn_wmma_f32_16x16x32_bf16(false, af, false, bf1,
                                                           (short)0, acc[sm][1], false, false);
    }
  };

  const int ksteps = Kd >> 5;
  stage(0, 0);
  wait_async0();
  __syncthreads();
  int cur = 0;
  for (int kt = 0; kt < ksteps; ++kt){
    int nxt = cur ^ 1;
    if (kt + 1 < ksteps) stage(kt + 1, nxt);
    compute(cur);
    wait_async0();
    __syncthreads();
    cur = nxt;
  }
#pragma unroll
  for (int sm = 0; sm < 2; ++sm){
#pragma unroll
    for (int sn = 0; sn < 2; ++sn){
      int n = n0 + ng*32 + sn*16 + lm;
#pragma unroll
      for (int j = 0; j < 8; ++j){
        int m = m0 + mg*32 + sm*16 + lh*8 + j;
        E[((long)b*C + m)*C + n] = acc[sm][sn][j];
      }
    }
  }
}

// ---------- CAM softmax: att = exp(minE - E)/sum  (== softmax(max-E)) ----------
__global__ void k_cam_softmax(const float* __restrict__ E, unsigned short* __restrict__ att, int C){
  int bc = blockIdx.x;
  const float* row = E + (long)bc*C;
  int tid = threadIdx.x;             // 256, C = 512
  __shared__ float red[256];
  float e0 = row[tid], e1 = row[tid + 256];
  red[tid] = fminf(e0, e1); __syncthreads();
  for (int st = 128; st > 0; st >>= 1){ if (tid < st) red[tid] = fminf(red[tid], red[tid+st]); __syncthreads(); }
  float mn = red[0]; __syncthreads();
  float p0 = __expf(mn - e0), p1 = __expf(mn - e1);
  red[tid] = p0 + p1; __syncthreads();
  for (int st = 128; st > 0; st >>= 1){ if (tid < st) red[tid] += red[tid+st]; __syncthreads(); }
  float inv = 1.f / red[0];
  unsigned short* o = att + (long)bc*C;
  o[tid]       = __builtin_bit_cast(unsigned short, f2bf(p0*inv));
  o[tid + 256] = __builtin_bit_cast(unsigned short, f2bf(p1*inv));
}

// ---------- CCA energies + softmax ----------
__global__ void k_cca_att(const unsigned short* __restrict__ qr, const unsigned short* __restrict__ kr,
                          float* __restrict__ att){
  const __bf16* q  = (const __bf16*)qr;
  const __bf16* kk = (const __bf16*)kr;
  int idx = blockIdx.x;              // b*H*W + h*W + w
  int w = idx % WW, h = (idx/WW) % HH, b = idx/(HH*WW);
  int tid = threadIdx.x;             // 192
  __shared__ float qv[16];
  __shared__ float red[256];
  if (tid < 16) qv[tid] = bf2f(q[((long)(b*16 + tid)*HH + h)*WW + w]);
  if (tid < 64) red[192 + tid] = -3e38f;
  __syncthreads();
  float e = 0.f;
  if (tid < 96){
    if (tid == h) e = -3e38f;
    else { for (int c = 0; c < 16; ++c) e += qv[c]*bf2f(kk[((long)(b*16 + c)*HH + tid)*WW + w]); }
  } else {
    int s = tid - 96;
    for (int c = 0; c < 16; ++c) e += qv[c]*bf2f(kk[((long)(b*16 + c)*HH + h)*WW + s]);
  }
  red[tid] = e; __syncthreads();
  for (int st = 128; st > 0; st >>= 1){ if (tid < st) red[tid] = fmaxf(red[tid], red[tid+st]); __syncthreads(); }
  float m = red[0]; __syncthreads();
  float p = __expf(e - m);
  red[tid] = p; if (tid < 64) red[192 + tid] = 0.f; __syncthreads();
  for (int st = 128; st > 0; st >>= 1){ if (tid < st) red[tid] += red[tid+st]; __syncthreads(); }
  att[(long)idx*192 + tid] = p / red[0];
}

// ---------- CCA aggregation + residual ----------
__global__ void k_cca_agg(const unsigned short* __restrict__ vr, const float* __restrict__ att,
                          const unsigned short* __restrict__ x1r, const float* __restrict__ gamma,
                          unsigned short* __restrict__ outr){
  const __bf16* v  = (const __bf16*)vr;
  const __bf16* x1 = (const __bf16*)x1r;
  int idx = blockIdx.x;              // (b*128 + c)*HH + h
  int h = idx % HH, c = (idx/HH) % 128, b = idx/(HH*128);
  int w = threadIdx.x;               // 96
  const float* a = att + ((long)(b*HH + h)*WW + w)*192;
  const __bf16* vb = v + (long)(b*128 + c)*HH*WW;
  float sum = 0.f;
  for (int s = 0; s < HH; ++s) sum += bf2f(vb[s*WW + w]) * a[s];
  for (int s = 0; s < WW; ++s) sum += bf2f(vb[h*WW + s]) * a[96 + s];
  long oi = ((long)(b*128 + c)*HH + h)*WW + w;
  float o = gamma[0]*sum + bf2f(x1[oi]);
  outr[oi] = __builtin_bit_cast(unsigned short, f2bf(o));
}

// ---------- final: out = bott2 + gamma_cam*cam_out + x ----------
__global__ void k_final(const float* __restrict__ bott2, const float* __restrict__ cam,
                        const float* __restrict__ x, const float* __restrict__ gamma,
                        float* __restrict__ out, int n){
  int i = blockIdx.x*256 + threadIdx.x;
  if (i < n) out[i] = bott2[i] + gamma[0]*cam[i] + x[i];
}

extern "C" void kernel_launch(void* const* d_in, const int* in_sizes, int n_in,
                              void* d_out, int out_size, void* d_ws, size_t ws_size,
                              hipStream_t stream){
  const float* x       = (const float*)d_in[0];
  const float* conva_w = (const float*)d_in[1];
  const float* bn1_s = (const float*)d_in[2];  const float* bn1_b = (const float*)d_in[3];
  const float* bn1_m = (const float*)d_in[4];  const float* bn1_v = (const float*)d_in[5];
  const float* wq = (const float*)d_in[6];     const float* bq = (const float*)d_in[7];
  const float* wk = (const float*)d_in[8];     const float* bk = (const float*)d_in[9];
  const float* wv = (const float*)d_in[10];    const float* bv = (const float*)d_in[11];
  const float* gcca   = (const float*)d_in[12];
  const float* convb_w= (const float*)d_in[13];
  const float* bn2_s = (const float*)d_in[14]; const float* bn2_b = (const float*)d_in[15];
  const float* bn2_m = (const float*)d_in[16]; const float* bn2_v = (const float*)d_in[17];
  const float* bot_w1 = (const float*)d_in[18];
  const float* bn3_s = (const float*)d_in[19]; const float* bn3_b = (const float*)d_in[20];
  const float* bn3_m = (const float*)d_in[21]; const float* bn3_v = (const float*)d_in[22];
  const float* bot_w2 = (const float*)d_in[23]; const float* bot_b2 = (const float*)d_in[24];
  const float* gcam   = (const float*)d_in[25];
  float* out = (float*)d_out;

  const int C = 512, IC = 128, QC = 16, B = 4;
  const long HW = (long)HH*WW;           // 9216

  char* p = (char*)d_ws;
  auto alloc = [&](size_t n)->char*{ char* r = p; p += (n + 255) & ~(size_t)255; return r; };

  unsigned short* xb     = (unsigned short*)alloc((size_t)B*C*HW*2);
  unsigned short* wa_b   = (unsigned short*)alloc((size_t)IC*C*9*2);
  unsigned short* out1   = (unsigned short*)alloc((size_t)B*IC*HW*2);
  unsigned short* wq_b   = (unsigned short*)alloc((size_t)QC*IC*2);
  unsigned short* wk_b   = (unsigned short*)alloc((size_t)QC*IC*2);
  unsigned short* wv_b   = (unsigned short*)alloc((size_t)IC*IC*2);
  unsigned short* qb     = (unsigned short*)alloc((size_t)B*QC*HW*2);
  unsigned short* kb2    = (unsigned short*)alloc((size_t)B*QC*HW*2);
  unsigned short* vb2    = (unsigned short*)alloc((size_t)B*IC*HW*2);
  float*          attcca = (float*)alloc((size_t)B*HW*192*4);
  unsigned short* ccao   = (unsigned short*)alloc((size_t)B*IC*HW*2);
  unsigned short* wb_b   = (unsigned short*)alloc((size_t)IC*IC*9*2);
  unsigned short* out2   = (unsigned short*)alloc((size_t)B*IC*HW*2);
  unsigned short* w1_b   = (unsigned short*)alloc((size_t)C*(C+IC)*9*2);
  unsigned short* bott1  = (unsigned short*)alloc((size_t)B*C*HW*2);
  unsigned short* w2_b   = (unsigned short*)alloc((size_t)C*C*2);
  float*          bott2  = (float*)alloc((size_t)B*C*HW*4);
  float*          energy = (float*)alloc((size_t)B*C*C*4);
  unsigned short* attcam = (unsigned short*)alloc((size_t)B*C*C*2);
  float*          camo   = (float*)alloc((size_t)B*C*HW*4);
  float* sc1 = (float*)alloc(IC*4); float* bi1 = (float*)alloc(IC*4);
  float* sc2 = (float*)alloc(IC*4); float* bi2 = (float*)alloc(IC*4);
  float* sc3 = (float*)alloc(C*4);  float* bi3 = (float*)alloc(C*4);

  auto CVT = [&](const float* s, unsigned short* d, long n){
    k_cvt_bf16<<<dim3((unsigned)((n + 255)/256)), dim3(256), 0, stream>>>(s, d, (int)n);
  };
  CVT(x, xb, (long)B*C*HW);
  CVT(conva_w, wa_b, (long)IC*C*9);
  CVT(wq, wq_b, QC*IC);
  CVT(wk, wk_b, QC*IC);
  CVT(wv, wv_b, IC*IC);
  CVT(convb_w, wb_b, (long)IC*IC*9);
  CVT(bot_w1, w1_b, (long)C*(C+IC)*9);
  CVT(bot_w2, w2_b, (long)C*C);

  k_bn_fuse<<<dim3(1), dim3(256), 0, stream>>>(bn1_s, bn1_b, bn1_m, bn1_v, sc1, bi1, IC);
  k_bn_fuse<<<dim3(1), dim3(256), 0, stream>>>(bn2_s, bn2_b, bn2_m, bn2_v, sc2, bi2, IC);
  k_bn_fuse<<<dim3(2), dim3(256), 0, stream>>>(bn3_s, bn3_b, bn3_m, bn3_v, sc3, bi3, C);

  dim3 blk(256);
  const unsigned NT = 3*(HH/4)*B;   // 288 N-tiles of 128 (4 rows x 32 x)

  // conva + bn1 + relu  (512 -> 128, 3x3)
  k_conv_wmma<3><<<dim3(NT, 1), blk, 0, stream>>>(xb, 512, nullptr, 0, wa_b, 0,
      128, 4608, sc1, bi1, 1, out1, nullptr);
  // q, k, v (1x1 + bias)
  k_conv_wmma<1><<<dim3(NT, 1), blk, 0, stream>>>(out1, 128, nullptr, 0, wq_b, 0,
      16, 128, nullptr, bq, 0, qb, nullptr);
  k_conv_wmma<1><<<dim3(NT, 1), blk, 0, stream>>>(out1, 128, nullptr, 0, wk_b, 0,
      16, 128, nullptr, bk, 0, kb2, nullptr);
  k_conv_wmma<1><<<dim3(NT, 1), blk, 0, stream>>>(out1, 128, nullptr, 0, wv_b, 0,
      128, 128, nullptr, bv, 0, vb2, nullptr);
  // criss-cross attention
  k_cca_att<<<dim3((unsigned)(B*HW)), dim3(192), 0, stream>>>(qb, kb2, attcca);
  k_cca_agg<<<dim3((unsigned)(B*128*HH)), dim3(96), 0, stream>>>(vb2, attcca, out1, gcca, ccao);
  // convb + bn2 + relu  (128 -> 128, 3x3)
  k_conv_wmma<3><<<dim3(NT, 1), blk, 0, stream>>>(ccao, 128, nullptr, 0, wb_b, 0,
      128, 1152, sc2, bi2, 1, out2, nullptr);
  // bottleneck conv1 on virtual concat [x, out2] (640 -> 512, 3x3) + bn3 + relu
  k_conv_wmma<3><<<dim3(NT, 4), blk, 0, stream>>>(xb, 512, out2, 128, w1_b, 0,
      512, 5760, sc3, bi3, 1, bott1, nullptr);
  // bottleneck conv2 (512 -> 512, 1x1) + bias -> f32
  k_conv_wmma<1><<<dim3(NT, 4), blk, 0, stream>>>(bott1, 512, nullptr, 0, w2_b, 0,
      512, 512, nullptr, bot_b2, 0, nullptr, bott2);
  // CAM: energy = X X^T (async-staged WMMA), softmax, out = att X (WMMA via conv kernel)
  k_gemm_abt<<<dim3(8, 4, 4), blk, 0, stream>>>(xb, energy, 512, (int)HW);
  k_cam_softmax<<<dim3((unsigned)(B*C)), dim3(256), 0, stream>>>(energy, attcam, C);
  k_conv_wmma<1><<<dim3(NT, 4), blk, 0, stream>>>(xb, 512, nullptr, 0, attcam, (long)C*C,
      512, 512, nullptr, nullptr, 0, nullptr, camo);
  // final residual combine
  int ntot = (int)(B*C*HW);
  k_final<<<dim3((unsigned)((ntot + 255)/256)), blk, 0, stream>>>(bott2, camo, x, gcam, out, ntot);
}
